// RecurrentWrapper_65695819759855
// MI455X (gfx1250) — matrix operand order repaired and checked by reading, not compiled, not run on hardware
//
#include <hip/hip_runtime.h>
#include <hip/hip_bf16.h>
#include <hip/hip_fp16.h>

// ---------------------------------------------------------------------------
// RecurrentWrapper for MI455X (gfx1250, wave32, WMMA)
//   proj = relu(x[B,T,D] @ W[D,H] + b)   -> f16 WMMA GEMM, f32 accumulate
//   h_t  = sigmoid(h_{t-1} + proj_t)     -> per-(b,h) sequential scan, in-place
// B=64 T=512 D=512 H=1024  => GEMM M=32768 K=512 N=1024 (34.4 GFLOP)
// f16 single-pass WMMA ~= HBM roofline (both ~14us); split-precision would be
// compute-bound, so we deliberately use v_wmma_f32_16x16x32_f16.
// ---------------------------------------------------------------------------

typedef __attribute__((ext_vector_type(16))) _Float16 v16h;
typedef __attribute__((ext_vector_type(8)))  _Float16 v8h;
typedef __attribute__((ext_vector_type(4)))  _Float16 v4h;
typedef __attribute__((ext_vector_type(8)))  float    v8f;
typedef __attribute__((ext_vector_type(4)))  float    v4f;

namespace cfg {
constexpr int B = 64, T = 512, D = 512, H = 1024;
constexpr int M = B * T;                 // 32768 GEMM rows
constexpr int BM = 128, BN = 64, BK = 32; // block tile
}

union V16U { v16h v; v8h h[2]; };

// ---------------------------------------------------------------------------
// Kernel 1: proj = relu(x @ W + b), written to `out` (d_out) as scratch.
// 256 threads = 8 waves arranged 4(M) x 2(N); each wave -> 32x32 = 2x2 WMMA.
// ---------------------------------------------------------------------------
__global__ __launch_bounds__(256)
void recurrent_proj_wmma_kernel(const float* __restrict__ x,
                                const float* __restrict__ W,
                                const float* __restrict__ bias,
                                float* __restrict__ out)
{
    using namespace cfg;
    __shared__ _Float16 As[BM * BK];  // A tile, row-major  [M][K]
    __shared__ _Float16 Bt[BN * BK];  // B tile, TRANSPOSED [N][K] so B-frag
                                      // gathers are contiguous b128 LDS loads
    const int tid  = threadIdx.x;
    const int lane = tid & 31;        // wave32
    const int wave = tid >> 5;
    const int wm   = wave >> 1;       // 0..3  -> M offset wm*32
    const int wn   = wave & 1;        // 0..1  -> N offset wn*32
    const int grp  = lane >> 4;       // lane half selects K-half of fragments
    const int l16  = lane & 15;

    const int m0 = blockIdx.x * BM;
    const int n0 = blockIdx.y * BN;

    v8f acc[2][2] = {};               // f32 accumulators (C/D, 8 VGPRs each)

    for (int k0 = 0; k0 < D; k0 += BK) {
        // ---- stage A: 128x32 f32 -> f16, row-major ----
        {
            const int kq = (tid & 7) * 4;          // 0,4,...,28
            int m = tid >> 3;                      // 0..31, +32 per rep
            #pragma unroll
            for (int r = 0; r < 4; ++r, m += 32) {
                v4f a = *(const v4f*)(x + (size_t)(m0 + m) * D + k0 + kq);
                v4h h; h.x = (_Float16)a.x; h.y = (_Float16)a.y;
                       h.z = (_Float16)a.z; h.w = (_Float16)a.w;
                *(v4h*)&As[m * BK + kq] = h;
            }
            if (k0 + BK < D) // emits global_prefetch_b8 for the next A tile
                __builtin_prefetch(x + (size_t)(m0 + (tid >> 3)) * D + k0 + BK, 0, 1);
        }
        // ---- stage B: 32x64 f32 -> f16, transposed into [N][K] ----
        {
            const int nq = (tid & 15) * 4;         // 0,4,...,60
            int kk = tid >> 4;                     // 0..15, +16 per rep
            #pragma unroll
            for (int r = 0; r < 2; ++r, kk += 16) {
                v4f w = *(const v4f*)(W + (size_t)(k0 + kk) * H + n0 + nq);
                Bt[(nq + 0) * BK + kk] = (_Float16)w.x;
                Bt[(nq + 1) * BK + kk] = (_Float16)w.y;
                Bt[(nq + 2) * BK + kk] = (_Float16)w.z;
                Bt[(nq + 3) * BK + kk] = (_Float16)w.w;
            }
        }
        __syncthreads();

        // ---- fragments per documented 16-bit layouts (CDNA5 ISA) ----
        // A 16x32: lane -> M=l16; elems 0-7 = K grp*8+0..7, elems 8-15 = 16+grp*8+0..7
        v16h afrag[2], bfrag[2];
        #pragma unroll
        for (int i = 0; i < 2; ++i) {
            const _Float16* ap = &As[(wm * 32 + i * 16 + l16) * BK + grp * 8];
            V16U t; t.h[0] = *(const v8h*)ap; t.h[1] = *(const v8h*)(ap + 16);
            afrag[i] = t.v;
        }
        // B 32x16: lane -> N=l16; elems 0-7 = K grp*8+0..7, elems 8-15 = 16+grp*8+0..7
        #pragma unroll
        for (int j = 0; j < 2; ++j) {
            const _Float16* bp = &Bt[(wn * 32 + j * 16 + l16) * BK + grp * 8];
            V16U t; t.h[0] = *(const v8h*)bp; t.h[1] = *(const v8h*)(bp + 16);
            bfrag[j] = t.v;
        }

        #pragma unroll
        for (int i = 0; i < 2; ++i)
            #pragma unroll
            for (int j = 0; j < 2; ++j)
                acc[i][j] = __builtin_amdgcn_wmma_f32_16x16x32_f16(
                    /*neg_a=*/false, afrag[i], /*neg_b=*/false, bfrag[j],
                    /*c_mod=*/(short)0, acc[i][j],
                    /*reuse_a=*/false, /*reuse_b=*/false);
        __syncthreads();
    }

    // ---- epilogue: bias + relu; C layout: elem r -> M = grp*8 + r, N = l16 ----
    #pragma unroll
    for (int i = 0; i < 2; ++i) {
        #pragma unroll
        for (int j = 0; j < 2; ++j) {
            const int n  = n0 + wn * 32 + j * 16 + l16;
            const float bb = bias[n];
            const int mb = m0 + wm * 32 + i * 16 + grp * 8;
            #pragma unroll
            for (int r = 0; r < 8; ++r) {
                float v = acc[i][j][r] + bb;
                out[(size_t)(mb + r) * H + n] = fmaxf(v, 0.0f);
            }
        }
    }
}

// ---------------------------------------------------------------------------
// Kernel 2: in-place scan over T. One thread per (b,h) chain (65536 chains).
// Reads proj_t, writes h_t to the same slot (each element read once, then
// overwritten -> deterministic across graph replays).
// ---------------------------------------------------------------------------
__global__ __launch_bounds__(256)
void recurrent_scan_kernel(float* __restrict__ out)
{
    using namespace cfg;
    const int u = blockIdx.x * blockDim.x + threadIdx.x; // b*H + h
    const int b = u / H, h = u % H;
    float* p = out + (size_t)b * T * H + h;

    float s = 0.0f;                                      // h0 = 0
    #pragma unroll 4
    for (int t = 0; t < T; ++t) {
        const float z = s + p[(size_t)t * H];
        // sigmoid via v_exp_f32 + v_rcp_f32
        s = __builtin_amdgcn_rcpf(1.0f + __expf(-z));
        __builtin_nontemporal_store(s, p + (size_t)t * H); // result not re-read
    }
}

// ---------------------------------------------------------------------------
extern "C" void kernel_launch(void* const* d_in, const int* in_sizes, int n_in,
                              void* d_out, int out_size, void* d_ws, size_t ws_size,
                              hipStream_t stream)
{
    (void)in_sizes; (void)n_in; (void)out_size; (void)d_ws; (void)ws_size;
    using namespace cfg;
    const float* x    = (const float*)d_in[0];
    const float* W    = (const float*)d_in[1];
    const float* bias = (const float*)d_in[2];
    float* out = (float*)d_out;

    dim3 grid1(M / BM, H / BN);                 // 256 x 16 blocks
    recurrent_proj_wmma_kernel<<<grid1, 256, 0, stream>>>(x, W, bias, out);

    recurrent_scan_kernel<<<(B * H) / 256, 256, 0, stream>>>(out);
}
